// YarnixLM_25477746000457
// MI455X (gfx1250) — matrix operand order repaired and checked by reading, not compiled, not run on hardware
//
#include <hip/hip_runtime.h>
#include <hip/hip_bf16.h>
#include <math.h>

// Problem dims (match reference)
#define Vv 512
#define Ee 128
#define Hh 512
#define Bb 64
#define Tt 2048
#define QSs 0.05f

typedef __attribute__((ext_vector_type(16))) __bf16        v16bf;
typedef __attribute__((ext_vector_type(8)))  float         v8f;
typedef __attribute__((ext_vector_type(4)))  unsigned int  u32x4;

// LDS row pitches in bf16 elements (pad 16B to spread banks)
#define XA_PITCH 136   // 128 + 8
#define HP       520   // 512 + 8

union AFrag { u32x4 u[2]; v16bf v; };

__device__ __forceinline__ v8f wmma_bf16(v16bf a, v16bf b, v8f c) {
  // D = A(16x32 bf16) * B(32x16 bf16) + C(16x16 f32)
  return __builtin_amdgcn_wmma_f32_16x16x32_bf16(false, a, false, b, (short)0, c,
                                                 false, false);
}

// A fragment (16x32 bf16) from LDS, row-major [16][pitch].
// Lane l (l&15 = M row): dwords 0-3 hold K = k + (l>>4)*8 + [0,8),
// dwords 4-7 hold K = k + 16 + (l>>4)*8 + [0,8).
__device__ __forceinline__ v16bf frag_a_lds(const __bf16* base, int pitch,
                                            int row, int grp, int k) {
  const __bf16* p = base + row * pitch + k + grp * 8;
  AFrag f;
  f.u[0] = *(const u32x4*)(p);
  f.u[1] = *(const u32x4*)(p + 16);
  return f.v;
}

// B fragment (32x16 bf16) from global, pre-transposed weight Wt[n][k] (row-major over K).
// Lane l holds column N = col, K = k + (l>>4)*8 + [0,8) then k + 16 + (l>>4)*8 + [0,8).
__device__ __forceinline__ v16bf frag_b_glb(const __bf16* Wt, int K,
                                            int col, int grp, int k) {
  const __bf16* p = Wt + (size_t)col * K + k + grp * 8;
  AFrag f;
  f.u[0] = *(const u32x4*)(p);
  f.u[1] = *(const u32x4*)(p + 16);
  return f.v;
}

// One GEMM phase over the K dimension for a pair of 16x16 output tiles that
// share the A operand. Software-pipelined one iteration deep: the k+32
// fragments are requested before the k WMMAs issue, so each wave keeps a full
// iteration of loads (2x global_b128 x2 tiles + 2x ds_b128) in flight behind
// the matrix pipe instead of wait(0)-ing on just-issued loads.
template <int K>
__device__ __forceinline__ void gemm_phase(const __bf16* alds, int apitch,
                                           const __bf16* wt, int c0, int c1,
                                           int row, int grp, v8f& acc0, v8f& acc1) {
  v16bf a  = frag_a_lds(alds, apitch, row, grp, 0);
  v16bf b0 = frag_b_glb(wt, K, c0, grp, 0);
  v16bf b1 = frag_b_glb(wt, K, c1, grp, 0);
#pragma unroll
  for (int k = 0; k < K; k += 32) {
    v16bf ac = a, b0c = b0, b1c = b1;
    if (k + 32 < K) {
      a  = frag_a_lds(alds, apitch, row, grp, k + 32);
      b0 = frag_b_glb(wt, K, c0, grp, k + 32);
      b1 = frag_b_glb(wt, K, c1, grp, k + 32);
    }
    acc0 = wmma_bf16(ac, b0c, acc0);
    acc1 = wmma_bf16(ac, b1c, acc1);
  }
}

// Fast tanh: clamp + exp-based rational (v_exp_f32 + fast divide).
__device__ __forceinline__ float fast_tanh(float x) {
  x = fmaxf(-15.f, fminf(15.f, x));
  float e = __expf(2.f * x);
  return __fdividef(e - 1.f, e + 1.f);
}

// Elementwise cell update for one 16x16 tile column (8 rows per lane).
__device__ __forceinline__ void cell_update(v8f drive, float freq, const float* amp,
                                            float clk, float* ang, float* hst,
                                            __bf16* hlds, int col, int grp) {
#pragma unroll
  for (int r = 0; r < 8; ++r) {
    float d = drive[r];
    float a = ang[r] + freq * fast_tanh(d);
    ang[r] = a;
    float osc = amp[0] * __sinf(a) + amp[1] * __sinf(2.f * a) +
                amp[2] * __sinf(4.f * a) + amp[3] * __sinf(8.f * a);
    float c = fast_tanh(d + osc);
    c = (1.f - QSs) * c + QSs * (rintf(c * 8.f) * 0.125f);
    float h = clk * hst[r] + (1.f - clk) * c;
    hst[r] = h;
    hlds[(grp * 8 + r) * HP + col] = (__bf16)h;
  }
}

// f32 -> bf16 transpose: dst[n][k] = src[k][n], src is K x N row-major.
__global__ void xpose_bf16(const float* __restrict__ src, __bf16* __restrict__ dst,
                           int K, int N) {
  int i = blockIdx.x * blockDim.x + threadIdx.x;
  if (i >= K * N) return;
  int n = i / K, k = i - n * K;
  dst[(size_t)n * K + k] = (__bf16)src[(size_t)k * N + n];
}

// Persistent recurrent kernel: 4 blocks (one 16-row batch tile each), 512 threads
// = 16 waves, 2 column tiles (32 cols) per wave. Fused readout.
__global__ __launch_bounds__(512) void yarnix_rnn(
    const int* __restrict__ x, const float* __restrict__ emb,
    const __bf16* __restrict__ Wti0, const __bf16* __restrict__ Wtr0,
    const float* __restrict__ b0, const float* __restrict__ f0,
    const float* __restrict__ a0,
    const __bf16* __restrict__ Wti1, const __bf16* __restrict__ Wtr1,
    const float* __restrict__ b1, const float* __restrict__ f1,
    const float* __restrict__ a1,
    const __bf16* __restrict__ Wto, const float* __restrict__ bo,
    float* __restrict__ out) {
  __shared__ __bf16 xa[16 * XA_PITCH];
  __shared__ __bf16 h0l[16 * HP];
  __shared__ __bf16 h1l[16 * HP];

  const int tid = threadIdx.x;
  const int wv = tid >> 5, ln = tid & 31;
  const int row = ln & 15, grp = ln >> 4;
  const int bb = blockIdx.x * 16;
  const int n0 = wv * 32;
  const int c0 = n0 + row, c1 = n0 + 16 + row;

  const float CLKS[4] = {0.5f, 0.8f, 0.95f, 0.999f};

  // Per-lane, per-column constants (static tile assignment for all T steps)
  float bi0[2] = {b0[c0], b0[c1]};
  float bi1[2] = {b1[c0], b1[c1]};
  float bio[2] = {bo[c0], bo[c1]};
  float fr0[2] = {f0[c0], f0[c1]};
  float fr1[2] = {f1[c0], f1[c1]};
  float clkv[2] = {CLKS[c0 >> 7], CLKS[c1 >> 7]};  // H/4 = 128
  float am0[2][4], am1[2][4];
#pragma unroll
  for (int k = 0; k < 4; ++k) {
    am0[0][k] = a0[k * Hh + c0];
    am0[1][k] = a0[k * Hh + c1];
    am1[0][k] = a1[k * Hh + c0];
    am1[1][k] = a1[k * Hh + c1];
  }

  // Persistent per-lane f32 state: 8 rows x 2 tiles x 2 layers (h and angle)
  float ang0a[8] = {}, ang0b[8] = {}, ang1a[8] = {}, ang1b[8] = {};
  float h0a[8] = {}, h0b[8] = {}, h1a[8] = {}, h1b[8] = {};

  for (int i = tid; i < 16 * HP; i += 512) {
    h0l[i] = (__bf16)0.f;
    h1l[i] = (__bf16)0.f;
  }
  __syncthreads();

  for (int t = 0; t < Tt; ++t) {
    // Anti-LICM: launder a scalar byte-offset (NOT the pointers) through a
    // volatile asm each iteration so the loop-invariant weight-fragment loads
    // cannot be hoisted (round 1 spilled ~4KB/lane of hoisted weights to
    // scratch), while pointer provenance survives and codegen keeps the
    // global_load path (round 2's pointer laundering degraded to flat_load).
    size_t opq = 0;
    asm volatile("" : "+s"(opq));
    const __bf16* wti0 = Wti0 + opq;
    const __bf16* wtr0 = Wtr0 + opq;
    const __bf16* wti1 = Wti1 + opq;
    const __bf16* wtr1 = Wtr1 + opq;
    const __bf16* wto  = Wto + opq;

    // ---- stage embedding tile (16 x 128) as bf16 into LDS ----
    {
      int r = tid >> 5;
      int cc = (tid & 31) * 4;
      int tok = x[(size_t)(bb + r) * Tt + t];
      float4 f4 = *(const float4*)(emb + (size_t)tok * Ee + cc);
      __bf16* dst = &xa[r * XA_PITCH + cc];
      dst[0] = (__bf16)f4.x;
      dst[1] = (__bf16)f4.y;
      dst[2] = (__bf16)f4.z;
      dst[3] = (__bf16)f4.w;
    }
    __syncthreads();

    // ---- layer 0: drive = x@Win0 + h0@Wrec0 + b0 ----
    v8f acc0, acc1;
#pragma unroll
    for (int r = 0; r < 8; ++r) { acc0[r] = bi0[0]; acc1[r] = bi0[1]; }
    gemm_phase<Ee>(xa, XA_PITCH, wti0, c0, c1, row, grp, acc0, acc1);
    gemm_phase<Hh>(h0l, HP, wtr0, c0, c1, row, grp, acc0, acc1);
    __syncthreads();  // all K reads of h0l complete before rewrite
    cell_update(acc0, fr0[0], am0[0], clkv[0], ang0a, h0a, h0l, c0, grp);
    cell_update(acc1, fr0[1], am0[1], clkv[1], ang0b, h0b, h0l, c1, grp);
    __syncthreads();  // h0l now holds h0_new

    // ---- layer 1: drive = h0_new@Win1 + h1@Wrec1 + b1 ----
#pragma unroll
    for (int r = 0; r < 8; ++r) { acc0[r] = bi1[0]; acc1[r] = bi1[1]; }
    gemm_phase<Hh>(h0l, HP, wti1, c0, c1, row, grp, acc0, acc1);
    gemm_phase<Hh>(h1l, HP, wtr1, c0, c1, row, grp, acc0, acc1);
    __syncthreads();
    cell_update(acc0, fr1[0], am1[0], clkv[0], ang1a, h1a, h1l, c0, grp);
    cell_update(acc1, fr1[1], am1[1], clkv[1], ang1b, h1b, h1l, c1, grp);
    __syncthreads();  // h1l now holds h1_new

    // ---- fused readout: logits[b, t, :] = h1_new @ W_out + b_out ----
#pragma unroll
    for (int r = 0; r < 8; ++r) { acc0[r] = bio[0]; acc1[r] = bio[1]; }
    gemm_phase<Hh>(h1l, HP, wto, c0, c1, row, grp, acc0, acc1);
    {
      size_t base = ((size_t)(bb + grp * 8) * Tt + t) * Vv;
#pragma unroll
      for (int r = 0; r < 8; ++r) {
        out[base + (size_t)r * Tt * Vv + c0] = acc0[r];
        out[base + (size_t)r * Tt * Vv + c1] = acc1[r];
      }
    }
    // No trailing barrier needed: next iteration's xa staging writes are only
    // read after the post-stage barrier, and h-buffer reads of this step all
    // completed before the post-L0/post-L1 barriers above.
  }
}

extern "C" void kernel_launch(void* const* d_in, const int* in_sizes, int n_in,
                              void* d_out, int out_size, void* d_ws, size_t ws_size,
                              hipStream_t stream) {
  const int*   x   = (const int*)d_in[0];
  const float* emb = (const float*)d_in[1];
  const float* Wi0 = (const float*)d_in[2];
  const float* Wr0 = (const float*)d_in[3];
  const float* b0  = (const float*)d_in[4];
  const float* f0  = (const float*)d_in[5];
  const float* a0  = (const float*)d_in[6];
  const float* Wi1 = (const float*)d_in[7];
  const float* Wr1 = (const float*)d_in[8];
  const float* b1  = (const float*)d_in[9];
  const float* f1  = (const float*)d_in[10];
  const float* a1  = (const float*)d_in[11];
  const float* Wo  = (const float*)d_in[12];
  const float* bo  = (const float*)d_in[13];

  // Workspace layout: transposed bf16 weights (~2.2 MB total)
  __bf16* Wti0 = (__bf16*)d_ws;            // [512][128]
  __bf16* Wtr0 = Wti0 + 512 * 128;         // [512][512]
  __bf16* Wti1 = Wtr0 + 512 * 512;         // [512][512]
  __bf16* Wtr1 = Wti1 + 512 * 512;         // [512][512]
  __bf16* Wto  = Wtr1 + 512 * 512;         // [512][512]

  const int thr = 256;
  xpose_bf16<<<(128 * 512 + thr - 1) / thr, thr, 0, stream>>>(Wi0, Wti0, 128, 512);
  xpose_bf16<<<(512 * 512 + thr - 1) / thr, thr, 0, stream>>>(Wr0, Wtr0, 512, 512);
  xpose_bf16<<<(512 * 512 + thr - 1) / thr, thr, 0, stream>>>(Wi1, Wti1, 512, 512);
  xpose_bf16<<<(512 * 512 + thr - 1) / thr, thr, 0, stream>>>(Wr1, Wtr1, 512, 512);
  xpose_bf16<<<(512 * 512 + thr - 1) / thr, thr, 0, stream>>>(Wo,  Wto,  512, 512);

  yarnix_rnn<<<4, 512, 0, stream>>>(x, emb, Wti0, Wtr0, b0, f0, a0,
                                    Wti1, Wtr1, b1, f1, a1, Wto, bo,
                                    (float*)d_out);
}
